// Mamba_87479893885120
// MI455X (gfx1250) — matrix operand order repaired and checked by reading, not compiled, not run on hardware
//
#include <hip/hip_runtime.h>
#include <hip/hip_bf16.h>

// ---------------- problem constants (match reference) ----------------
#define DMODEL   1024
#define DSTATE   16
#define DCONV    4
#define DINNER   2048          // EXPAND * DMODEL
#define DTRANK   64
#define NB       2
#define LSEQ     2048
#define NTOK     (NB * LSEQ)   // 4096 tokens
#define XPROJ_N  (DTRANK + 2 * DSTATE)   // 96

// ---------------- GEMM tiling ----------------
#define BM 128
#define BN 128
#define BK 32
#define PAD_LD 40              // bf16 elems per LDS row (32 + 8 pad, 80B stride)

// ---------------- WMMA vector types (CDNA5, wave32) ----------------
typedef __attribute__((ext_vector_type(16))) __bf16 v16bf;
typedef __attribute__((ext_vector_type(8)))  __bf16 v8bf;
typedef __attribute__((ext_vector_type(8)))  float  v8f;

// pointer types for the async global->LDS builtin:
// (v4i addrspace(1)*, v4i addrspace(3)*, imm offset, imm cpol)
typedef int v4i_vec __attribute__((vector_size(16)));
typedef __attribute__((address_space(1))) v4i_vec* gas_v4i;
typedef __attribute__((address_space(3))) v4i_vec* las_v4i;

// ---------------- CDNA5 async global->LDS path (guarded) ----------------
#if __has_builtin(__builtin_amdgcn_global_load_async_to_lds_b128)
  #define HAVE_ASYNC_LDS 1
#else
  #define HAVE_ASYNC_LDS 0
#endif

#if HAVE_ASYNC_LDS
  #if __has_builtin(__builtin_amdgcn_s_wait_asynccnt)
    #define WAIT_ASYNC(imm) __builtin_amdgcn_s_wait_asynccnt(imm)
  #else
    #define WAIT_ASYNC(imm) asm volatile("s_wait_asynccnt " #imm ::: "memory")
  #endif
#else
  #define WAIT_ASYNC(imm)
#endif

// =====================================================================
// f32 -> bf16 cast
// =====================================================================
__global__ __launch_bounds__(256)
void k_cast_bf16(const float* __restrict__ s, __bf16* __restrict__ d, int n) {
  int i = blockIdx.x * blockDim.x + threadIdx.x;
  if (i < n) d[i] = (__bf16)s[i];
}

// =====================================================================
// Cooperative tile fill: 128 rows x 32 k-elems (bf16) -> LDS.
// 512 16B chunks, 2 per thread. Out-of-range rows clamped to rowMax
// (duplicate data; corresponding output columns are never stored).
// =====================================================================
__device__ __forceinline__
void fill_tile(const __bf16* __restrict__ G, int ld, int row0, int k0,
               int rowMax, __bf16 (*__restrict__ S)[PAD_LD], int tid) {
#pragma unroll
  for (int it = 0; it < 2; ++it) {
    const int c  = tid + it * 256;      // chunk id 0..511
    const int r  = c >> 2;              // tile row 0..127
    const int kc = (c & 3) << 3;        // k element offset 0,8,16,24
    int rg = row0 + r;
    rg = (rg > rowMax) ? rowMax : rg;
    const __bf16* gp = G + (size_t)rg * ld + (k0 + kc);
    __bf16* lp = &S[r][kc];
#if HAVE_ASYNC_LDS
    __builtin_amdgcn_global_load_async_to_lds_b128(
        (gas_v4i)(v4i_vec*)gp,       // const-strip + reinterpret, then AS1
        (las_v4i)(v4i_vec*)lp,       // reinterpret, then AS3
        0, 0);
#else
    *(v8bf*)lp = *(const v8bf*)gp;
#endif
  }
}

// =====================================================================
// GEMM: C[M,N] (f32, ldc) = A[M,K] (bf16, lda) x B[N,K]^T (bf16, ldb)
// Requires M % 128 == 0, K % 32 == 0; N ragged OK (clamped loads,
// guarded stores). Block = 256 thr = 8 waves (4x2); block tile 128x128;
// wave tile 32x64 = 2x4 WMMA. Double-buffered LDS staging via
// GLOBAL_LOAD_ASYNC_TO_LDS_B128 (ASYNCcnt) when available.
// =====================================================================
__global__ __launch_bounds__(256)
void k_gemm_bf16(const __bf16* __restrict__ A, const __bf16* __restrict__ B,
                 float* __restrict__ C, int M, int N, int K,
                 int lda, int ldb, int ldc) {
  __shared__ __bf16 sA[2][BM][PAD_LD];   // 2 x 10 KB
  __shared__ __bf16 sB[2][BN][PAD_LD];   // 2 x 10 KB

  const int tid   = threadIdx.x;
  const int lane  = tid & 31;
  const int wave  = tid >> 5;
  const int waveM = wave >> 1;          // 0..3  (rows, 32 each)
  const int waveN = wave & 1;           // 0..1  (cols, 64 each)
  const int l15   = lane & 15;
  const int hi    = lane >> 4;          // half-wave select

  const int rowBlk = blockIdx.y * BM;
  const int colBlk = blockIdx.x * BN;

  v8f acc[2][4];
#pragma unroll
  for (int i = 0; i < 2; ++i)
#pragma unroll
    for (int j = 0; j < 4; ++j)
#pragma unroll
      for (int r = 0; r < 8; ++r) acc[i][j][r] = 0.0f;

  const int nk = K / BK;

  // prologue: stage 0
  fill_tile(A, lda, rowBlk, 0, M - 1, sA[0], tid);
  fill_tile(B, ldb, colBlk, 0, N - 1, sB[0], tid);

  for (int kt = 0; kt < nk; ++kt) {
    const int st = kt & 1;
    if (kt + 1 < nk) {
      // prefetch next stage (4 async b128 per thread)
      fill_tile(A, lda, rowBlk, (kt + 1) * BK, M - 1, sA[st ^ 1], tid);
      fill_tile(B, ldb, colBlk, (kt + 1) * BK, N - 1, sB[st ^ 1], tid);
      WAIT_ASYNC(4);   // current stage (issued last iter) complete
    } else {
      WAIT_ASYNC(0);
    }
    __syncthreads();

    // ---- fragments from LDS (padded rows -> conflict-free ds_load_b128)
    v16bf afrag[2], bfrag[4];
#pragma unroll
    for (int i = 0; i < 2; ++i) {
      // A 16x32 frag: lane<16 row=l15 K{0..7,16..23}; lane>=16 K{8..15,24..31}
      const __bf16* pa = &sA[st][waveM * 32 + i * 16 + l15][hi * 8];
      v8bf lo = *(const v8bf*)pa;
      v8bf up = *(const v8bf*)(pa + 16);
      afrag[i] = __builtin_shufflevector(lo, up,
                   0, 1, 2, 3, 4, 5, 6, 7, 8, 9, 10, 11, 12, 13, 14, 15);
    }
#pragma unroll
    for (int j = 0; j < 4; ++j) {
      // B 32x16 frag: col=l15, lane<16 K{0..7,16..23}; lane>=16 K{8..15,24..31}
      const __bf16* pb = &sB[st][waveN * 64 + j * 16 + l15][hi * 8];
      v8bf lo = *(const v8bf*)pb;
      v8bf up = *(const v8bf*)(pb + 16);
      bfrag[j] = __builtin_shufflevector(lo, up,
                   0, 1, 2, 3, 4, 5, 6, 7, 8, 9, 10, 11, 12, 13, 14, 15);
    }
    // ---- 2x4 matrix-core multiplies (fp32 accumulate)
#pragma unroll
    for (int i = 0; i < 2; ++i)
#pragma unroll
      for (int j = 0; j < 4; ++j)
        acc[i][j] = __builtin_amdgcn_wmma_f32_16x16x32_bf16(
            false, afrag[i], false, bfrag[j],
            (short)0, acc[i][j], false, false);

    __syncthreads();   // safe to overwrite stage st next iteration
  }

  // ---- store C: VGPR r -> row (hi*8 + r), col = l15
  const int mh = hi * 8;
#pragma unroll
  for (int i = 0; i < 2; ++i) {
#pragma unroll
    for (int j = 0; j < 4; ++j) {
      const int col = colBlk + waveN * 64 + j * 16 + l15;
      if (col < N) {
#pragma unroll
        for (int r = 0; r < 8; ++r) {
          const int row = rowBlk + waveM * 32 + i * 16 + mh + r;
          C[(size_t)row * ldc + col] = acc[i][j][r];
        }
      }
    }
  }
}

// =====================================================================
// Causal depthwise conv (width 4) + bias + SiLU on x = xz[..., :DINNER].
// Emits u in f32 (for scan) and bf16 (A operand of x_proj GEMM).
// =====================================================================
__global__ __launch_bounds__(256)
void k_conv_silu(const float* __restrict__ xz, const float* __restrict__ cw,
                 const float* __restrict__ cb, float* __restrict__ u,
                 __bf16* __restrict__ u_bf) {
  int idx = blockIdx.x * blockDim.x + threadIdx.x;
  if (idx >= NTOK * DINNER) return;
  const int d = idx % DINNER;
  const int t = (idx / DINNER) % LSEQ;
  const int b = idx / (DINNER * LSEQ);
  float acc = cb[d];
#pragma unroll
  for (int j = 0; j < DCONV; ++j) {
    const int ts = t - (DCONV - 1) + j;
    if (ts >= 0)
      acc += xz[(size_t)(b * LSEQ + ts) * (2 * DINNER) + d] * cw[d * DCONV + j];
  }
  const float s = acc / (1.0f + __expf(-acc));   // SiLU
  u[idx] = s;
  u_bf[idx] = (__bf16)s;
}

// =====================================================================
// Extract dt columns (0..63) of x_dbl (ld 96) into packed bf16 (ld 64).
// =====================================================================
__global__ __launch_bounds__(256)
void k_extract_dt(const float* __restrict__ xdbl, __bf16* __restrict__ dt_bf) {
  int idx = blockIdx.x * blockDim.x + threadIdx.x;
  if (idx >= NTOK * DTRANK) return;
  const int m = idx / DTRANK;
  const int r = idx % DTRANK;
  dt_bf[idx] = (__bf16)xdbl[(size_t)m * XPROJ_N + r];
}

// =====================================================================
// Selective scan. One state (b, d, n) per lane: 16 lanes per channel,
// 65536 threads = 2048 waves. Tree-reduce over the 16 states with
// __shfl_xor, then lane n==0 fuses +u*D, *silu(z), bf16 store of y.
// grid = (DINNER/16, NB), block = 256.
// =====================================================================
__global__ __launch_bounds__(256)
void k_scan(const float* __restrict__ xz, const float* __restrict__ u,
            const float* __restrict__ dlin, const float* __restrict__ xdbl,
            const float* __restrict__ dt_b, const float* __restrict__ A_log,
            const float* __restrict__ Dp, __bf16* __restrict__ y_bf) {
  const int tid  = threadIdx.x;
  const int lane = tid & 31;
  const int n    = lane & 15;          // state index
  const int half = lane >> 4;
  const int ch   = ((tid >> 5) << 1) | half;   // 0..15 channel within block
  const int d    = blockIdx.x * 16 + ch;
  const int b    = blockIdx.y;

  const float Aa  = -__expf(A_log[d * DSTATE + n]);  // A = -exp(A_log)
  const float dtb = dt_b[d];
  const float Dd  = Dp[d];
  float h = 0.0f;

  const size_t tokBase = (size_t)b * LSEQ;
  for (int t = 0; t < LSEQ; ++t) {
    const size_t tok = tokBase + t;
    const float dl  = dlin[tok * DINNER + d] + dtb;
    const float dlt = (dl > 20.0f) ? dl : log1pf(__expf(dl));  // softplus
    const float ut  = u[tok * DINNER + d];
    const float Bn  = xdbl[tok * XPROJ_N + DTRANK + n];
    const float Cn  = xdbl[tok * XPROJ_N + DTRANK + DSTATE + n];
    h = __expf(dlt * Aa) * h + dlt * Bn * ut;   // h = exp(dt*A)*h + dt*B*u
    float yv = h * Cn;
    yv += __shfl_xor(yv, 1);
    yv += __shfl_xor(yv, 2);
    yv += __shfl_xor(yv, 4);
    yv += __shfl_xor(yv, 8);
    if (n == 0) {
      const float zt = xz[tok * (2 * DINNER) + DINNER + d];
      const float sz = zt / (1.0f + __expf(-zt));            // silu(z)
      const float yy = (yv + ut * Dd) * sz;
      y_bf[tok * DINNER + d] = (__bf16)yy;
    }
  }
}

// =====================================================================
// launch
// =====================================================================
extern "C" void kernel_launch(void* const* d_in, const int* in_sizes, int n_in,
                              void* d_out, int out_size, void* d_ws, size_t ws_size,
                              hipStream_t stream) {
  (void)in_sizes; (void)n_in; (void)out_size; (void)ws_size;

  const float* hs    = (const float*)d_in[0];   // (B,L,DMODEL)
  const float* Win   = (const float*)d_in[1];   // (2*DINNER, DMODEL)
  const float* convw = (const float*)d_in[2];   // (DINNER,1,4)
  const float* convb = (const float*)d_in[3];   // (DINNER,)
  const float* Wx    = (const float*)d_in[4];   // (96, DINNER)
  const float* Wdt   = (const float*)d_in[5];   // (DINNER, 64)
  const float* bdt   = (const float*)d_in[6];   // (DINNER,)
  const float* Alog  = (const float*)d_in[7];   // (DINNER, 16)
  const float* Dp    = (const float*)d_in[8];   // (DINNER,)
  const float* Wout  = (const float*)d_in[9];   // (DMODEL, DINNER)
  float* out = (float*)d_out;                   // (B,L,DMODEL)

  // ---- carve workspace (256B aligned regions) ----
  char* ws = (char*)d_ws;
  size_t off = 0;
  auto take = [&](size_t bytes) -> void* {
    void* p = ws + off;
    off += (bytes + 255) & ~(size_t)255;
    return p;
  };
  __bf16* hid_bf  = (__bf16*)take((size_t)NTOK * DMODEL * 2);        // 8 MB
  __bf16* win_bf  = (__bf16*)take((size_t)2 * DINNER * DMODEL * 2);  // 8 MB
  float*  xz      = (float*) take((size_t)NTOK * 2 * DINNER * 4);    // 64 MB
  float*  u       = (float*) take((size_t)NTOK * DINNER * 4);        // 32 MB
  __bf16* u_bf    = (__bf16*)take((size_t)NTOK * DINNER * 2);        // 16 MB
  __bf16* wx_bf   = (__bf16*)take((size_t)XPROJ_N * DINNER * 2);
  float*  xdbl    = (float*) take((size_t)NTOK * XPROJ_N * 4);
  __bf16* dt_bf   = (__bf16*)take((size_t)NTOK * DTRANK * 2);
  __bf16* wdt_bf  = (__bf16*)take((size_t)DINNER * DTRANK * 2);
  float*  dlin    = (float*) take((size_t)NTOK * DINNER * 4);        // 32 MB
  __bf16* wout_bf = (__bf16*)take((size_t)DMODEL * DINNER * 2);      // 4 MB
  // y_bf (16 MB) reuses hid_bf + win_bf region (dead after in_proj GEMM)
  __bf16* y_bf = hid_bf;

  // ---- bf16 casts of activations & weights ----
  k_cast_bf16<<<(NTOK * DMODEL + 255) / 256, 256, 0, stream>>>(hs, hid_bf, NTOK * DMODEL);
  k_cast_bf16<<<(2 * DINNER * DMODEL + 255) / 256, 256, 0, stream>>>(Win, win_bf, 2 * DINNER * DMODEL);
  k_cast_bf16<<<(XPROJ_N * DINNER + 255) / 256, 256, 0, stream>>>(Wx, wx_bf, XPROJ_N * DINNER);
  k_cast_bf16<<<(DINNER * DTRANK + 255) / 256, 256, 0, stream>>>(Wdt, wdt_bf, DINNER * DTRANK);
  k_cast_bf16<<<(DMODEL * DINNER + 255) / 256, 256, 0, stream>>>(Wout, wout_bf, DMODEL * DINNER);

  // ---- in_proj: xz = hs @ Win^T  (4096 x 4096 x 1024) ----
  {
    dim3 g((2 * DINNER + BN - 1) / BN, (NTOK + BM - 1) / BM);
    k_gemm_bf16<<<g, 256, 0, stream>>>(hid_bf, win_bf, xz,
                                       NTOK, 2 * DINNER, DMODEL,
                                       DMODEL, DMODEL, 2 * DINNER);
  }
  // ---- causal depthwise conv + SiLU -> u ----
  k_conv_silu<<<(NTOK * DINNER + 255) / 256, 256, 0, stream>>>(xz, convw, convb, u, u_bf);

  // ---- x_proj: xdbl = u @ Wx^T  (4096 x 96 x 2048) ----
  {
    dim3 g((XPROJ_N + BN - 1) / BN, (NTOK + BM - 1) / BM);
    k_gemm_bf16<<<g, 256, 0, stream>>>(u_bf, wx_bf, xdbl,
                                       NTOK, XPROJ_N, DINNER,
                                       DINNER, DINNER, XPROJ_N);
  }
  // ---- dt_proj: dlin = dt @ Wdt^T  (4096 x 2048 x 64) ----
  k_extract_dt<<<(NTOK * DTRANK + 255) / 256, 256, 0, stream>>>(xdbl, dt_bf);
  {
    dim3 g((DINNER + BN - 1) / BN, (NTOK + BM - 1) / BM);
    k_gemm_bf16<<<g, 256, 0, stream>>>(dt_bf, wdt_bf, dlin,
                                       NTOK, DINNER, DTRANK,
                                       DTRANK, DTRANK, DINNER);
  }
  // ---- selective scan (+ skip, gate, bf16 y) ----
  {
    dim3 g(DINNER / 16, NB);
    k_scan<<<g, 256, 0, stream>>>(xz, u, dlin, xdbl, bdt, Alog, Dp, y_bf);
  }
  // ---- out_proj: out = y @ Wout^T  (4096 x 1024 x 2048) ----
  {
    dim3 g((DMODEL + BN - 1) / BN, (NTOK + BM - 1) / BM);
    k_gemm_bf16<<<g, 256, 0, stream>>>(y_bf, wout_bf, out,
                                       NTOK, DMODEL, DINNER,
                                       DINNER, DINNER, DMODEL);
  }
}
